// SpectralCOREL_52707838656552
// MI455X (gfx1250) — compile-verified, matrix operand order checked by reading, not run on hardware
//
#include <hip/hip_runtime.h>
#include <hip/hip_bf16.h>
#include <math.h>
#include <stdint.h>

// ---------------------------------------------------------------------------
// SpectralCOREL on MI455X (gfx1250, wave32, WMMA + TDM).
//  - Heavy GEMMs (edge-MLP layer 2, per-edge message matmuls) run on
//    v_wmma_f32_16x16x32_f16 with f16 operands staged K-contiguous so each
//    lane fragment is two aligned 16-byte loads (ISA 7.12.2 layout).
//  - Message kernel stages per-edge A (x[:,src], 128xK strided-2D) and B
//    (weT[e], 64xK contiguous) tiles into LDS via tensor_load_to_lds (TDM,
//    TENSORcnt) once per edge; 8 waves x 4 n-chunks consume them from LDS.
//  - Scatter-mean + pointwise tails stay fp32 VALU / global f32 atomics.
// ---------------------------------------------------------------------------

typedef _Float16 half_t;
typedef __attribute__((ext_vector_type(16))) _Float16 v16h;
typedef __attribute__((ext_vector_type(8)))  _Float16 v8h;
typedef __attribute__((ext_vector_type(8)))  float    v8f;
typedef __attribute__((ext_vector_type(4)))  unsigned int u32x4;
typedef __attribute__((ext_vector_type(8)))  int i32x8;
typedef __attribute__((ext_vector_type(4)))  int i32x4;

#define BB 128
#define NNODE 283
#define EE 8192
#define HH 64
#define K1PAD 32   // IN1=17 padded to 32 for one WMMA k-step

#if defined(__has_builtin)
#  if __has_builtin(__builtin_amdgcn_tensor_load_to_lds)
#    define HAVE_TDM 1
#  endif
#endif
#ifndef HAVE_TDM
#  define HAVE_TDM 0
#endif

#define RFL(x) ((unsigned)__builtin_amdgcn_readfirstlane((int)(x)))

// ISA 7.12.2 16-bit A/B fragment: lane l -> row (l&15); halves 0..7 hold
// K = kbase + (l>>4)*8 + 0..7, halves 8..15 hold K = kbase+16 + (l>>4)*8 + 0..7.
__device__ __forceinline__ v16h frag16(const half_t* row, int kbase, int hi) {
    v8h lo = *(const v8h*)(row + kbase + hi * 8);
    v8h hh = *(const v8h*)(row + kbase + 16 + hi * 8);
    v16h r;
#pragma unroll
    for (int i = 0; i < 8; ++i) { r[i] = lo[i]; r[i + 8] = hh[i]; }
    return r;
}

__device__ __forceinline__ v8f wmma_f16(v16h a, v16h b, v8f c) {
    return __builtin_amdgcn_wmma_f32_16x16x32_f16(false, a, false, b, (short)0, c, false, false);
}

#if HAVE_TDM
// Issue one 2D TDM tile load (global -> LDS). All sizes/strides in elements
// (data_size = 2 bytes). Descriptor per CDNA5 ISA 10.8 (D# groups 0/1).
__device__ __forceinline__ void tdm_issue_2d(unsigned lds_addr, const void* gptr,
                                             unsigned tile0, unsigned tile1,
                                             unsigned tdim0, unsigned tdim1,
                                             unsigned long long stride0) {
    unsigned long long ga = (unsigned long long)(uintptr_t)gptr;
    u32x4 g0;
    g0[0] = RFL(1u);                                       // count=1, user D#
    g0[1] = RFL(lds_addr);                                 // lds_addr [63:32]
    g0[2] = RFL((unsigned)(ga & 0xffffffffu));             // global_addr lo
    g0[3] = RFL(((unsigned)((ga >> 32) & 0x01ffffffu)) | (2u << 30)); // hi + type=2
    i32x8 g1;
    g1[0] = (int)RFL(1u << 16);                            // data_size=2B
    g1[1] = (int)RFL((tdim0 & 0xffffu) << 16);             // tensor_dim0 lo
    g1[2] = (int)RFL(((tdim0 >> 16) & 0xffffu) | ((tdim1 & 0xffffu) << 16));
    g1[3] = (int)RFL(((tdim1 >> 16) & 0xffffu) | ((tile0 & 0xffffu) << 16));
    g1[4] = (int)RFL(tile1 & 0xffffu);                     // tile_dim1, tile_dim2=0
    g1[5] = (int)RFL((unsigned)(stride0 & 0xffffffffu));   // dim0_stride lo
    g1[6] = (int)RFL((unsigned)((stride0 >> 32) & 0xffffu));
    g1[7] = 0;
    i32x4 z4 = {};
#if __clang_major__ >= 23
    i32x8 z8 = {};
    __builtin_amdgcn_tensor_load_to_lds(g0, g1, z4, z4, z8, 0);
#else
    __builtin_amdgcn_tensor_load_to_lds(g0, g1, z4, z4, 0);
#endif
}
#endif

// ---------------------------------------------------------------------- zero
__global__ void k_zero_u32(unsigned int* __restrict__ p, size_t n) {
    size_t i = (size_t)blockIdx.x * blockDim.x + threadIdx.x;
    if (i < n) p[i] = 0u;
}

// ------------------------------------------------- build xf[b][node][32] f16
__global__ void k_build_xf(const float* __restrict__ sigma, const float* __restrict__ posenc,
                           half_t* __restrict__ xf) {
    int t = blockIdx.x * blockDim.x + threadIdx.x;
    if (t >= BB * NNODE) return;
    int node = t % NNODE;
    half_t* row = xf + (size_t)t * K1PAD;
    row[0] = (half_t)sigma[t];
#pragma unroll
    for (int i = 0; i < 16; ++i) row[1 + i] = (half_t)posenc[node * 16 + i];
#pragma unroll
    for (int k = 17; k < K1PAD; ++k) row[k] = (half_t)0.0f;
}

// -------------------------------------- edge-MLP first layer: 3 -> 64 + relu
__global__ void k_hidden(const float* __restrict__ ea, const float* __restrict__ w1,
                         const float* __restrict__ b1, half_t* __restrict__ hid) {
    int t = blockIdx.x * blockDim.x + threadIdx.x;   // E*64 threads
    int e = t >> 6, o = t & 63;
    const float* a = ea + e * 3;
    float acc = b1[o] + a[0] * w1[o] + a[1] * w1[64 + o] + a[2] * w1[128 + o];
    hid[t] = (half_t)fmaxf(acc, 0.0f);
}

// ----------------------------- transpose+convert w2 [64][C] -> w2T [C][64] f16
__global__ void k_cvt_w2T(const float* __restrict__ w2, half_t* __restrict__ w2T, int C) {
    int t = blockIdx.x * blockDim.x + threadIdx.x;   // C*64 threads
    int c = t >> 6, k = t & 63;
    w2T[t] = (half_t)w2[(size_t)k * C + c];
}

// ------------- edge-MLP second layer via WMMA: hid[E,64] @ w2[64,C] (+b2),
// ------------- store transposed per-edge weights weT[e][o][kdim] f16.
__global__ void k_mlp_wmma(const half_t* __restrict__ hid, const half_t* __restrict__ w2T,
                           const float* __restrict__ b2, half_t* __restrict__ weT,
                           int C, int kdim) {
    int etile = blockIdx.x;              // 16 edges per tile
    int wave  = threadIdx.x >> 5;
    int lane  = threadIdx.x & 31;
    int chunk = blockIdx.y * 8 + wave;   // 16 output columns per chunk
    int nchunks = C >> 4;
    if (chunk >= nchunks) return;        // wave-uniform guard: EXEC stays all-1

    int idx = lane & 15, hi = lane >> 4;
    const half_t* arow = hid + (size_t)(etile * 16 + idx) * 64;
    const half_t* brow = w2T + (size_t)(chunk * 16 + idx) * 64;

    v8f c = {};
#pragma unroll
    for (int s = 0; s < 2; ++s)
        c = wmma_f16(frag16(arow, 32 * s, hi), frag16(brow, 32 * s, hi), c);

    int ncol = chunk * 16 + idx;         // column this lane owns
    float bias = b2[ncol];
    int i = ncol >> 6, o = ncol & 63;    // reshape(-1, IN, 64): col = i*64+o
#pragma unroll
    for (int v = 0; v < 8; ++v) {
        int e = etile * 16 + v + 8 * hi; // D row -> edge
        weT[((size_t)e * 64 + o) * kdim + i] = (half_t)(c[v] + bias);
    }
}

// ------------- per-edge message GEMM + atomic scatter into agg[b][node][64].
// ------------- Block = one edge; TDM stages A(128xK)/B(64xK) tiles to LDS;
// ------------- 8 waves (batch tiles) x 4 n-chunks of WMMA from LDS.
template <int KDIM, int KSTEPS>
__global__ void k_msg_wmma(const half_t* __restrict__ xf, const half_t* __restrict__ weT,
                           const int* __restrict__ src, const int* __restrict__ dst,
                           float* __restrict__ agg) {
    __shared__ __align__(16) half_t sA[128 * KDIM];
    __shared__ __align__(16) half_t sB[64 * KDIM];
    int e = blockIdx.x;
    int s = src[e], d = dst[e];

#if HAVE_TDM
    if ((threadIdx.x >> 5) == 0) {       // wave 0 issues both DMA descriptors
        // A: 128 batch rows of x[:, s, :], row stride NNODE*KDIM elements
        tdm_issue_2d((unsigned)(uintptr_t)&sA[0], xf + (size_t)s * KDIM,
                     KDIM, 128, KDIM, 128, (unsigned long long)NNODE * KDIM);
        // B: weT[e] = 64*KDIM contiguous elements
        tdm_issue_2d((unsigned)(uintptr_t)&sB[0], weT + (size_t)e * 64 * KDIM,
                     64 * KDIM, 1, 64 * KDIM, 1, (unsigned long long)64 * KDIM);
        __builtin_amdgcn_s_wait_tensorcnt(0);
    }
    __syncthreads();
#else
    {   // fallback: cooperative vectorized staging
        const int tid = threadIdx.x;
        for (int i = tid; i < 64 * KDIM / 8; i += 256)
            ((v8h*)sB)[i] = ((const v8h*)(weT + (size_t)e * 64 * KDIM))[i];
        for (int i = tid; i < 128 * KDIM / 8; i += 256) {
            int b = i / (KDIM / 8), w8 = i % (KDIM / 8);
            ((v8h*)sA)[i] = *(const v8h*)(xf + ((size_t)b * NNODE + s) * KDIM + w8 * 8);
        }
        __syncthreads();
    }
#endif

    int btile = threadIdx.x >> 5;        // 8 waves = 8 batch tiles of 16
    int lane  = threadIdx.x & 31;
    int idx = lane & 15, hi = lane >> 4;

    const half_t* arow = &sA[(btile * 16 + idx) * KDIM];
    v16h afr[KSTEPS];
#pragma unroll
    for (int ks = 0; ks < KSTEPS; ++ks) afr[ks] = frag16(arow, 32 * ks, hi);

#pragma unroll
    for (int nchunk = 0; nchunk < 4; ++nchunk) {
        const half_t* brow = &sB[(nchunk * 16 + idx) * KDIM];
        v8f c = {};
#pragma unroll
        for (int ks = 0; ks < KSTEPS; ++ks)
            c = wmma_f16(afr[ks], frag16(brow, 32 * ks, hi), c);
        int n = nchunk * 16 + idx;
#pragma unroll
        for (int v = 0; v < 8; ++v) {
            int b = btile * 16 + v + 8 * hi;
            atomicAdd(&agg[((size_t)b * NNODE + d) * 64 + n], c[v]);
        }
    }
}

// ----------------------------------------------------------- in-degree counts
__global__ void k_cnt(const int* __restrict__ dst, float* __restrict__ cnt) {
    int t = blockIdx.x * blockDim.x + threadIdx.x;
    if (t < EE) atomicAdd(&cnt[dst[t]], 1.0f);
}

// -------- layer1 finish: relu(x@root1 + agg/max(cnt,1) + bias1) -> xh (f16)
__global__ void k_combine1(const float* __restrict__ sigma, const float* __restrict__ posenc,
                           const float* __restrict__ root1, const float* __restrict__ bias1,
                           const float* __restrict__ agg1, const float* __restrict__ cnt,
                           half_t* __restrict__ xh) {
    int t = blockIdx.x * blockDim.x + threadIdx.x;   // B*N*64 threads
    int o = t & 63, bn = t >> 6;
    int node = bn % NNODE;
    float acc = bias1[o] + sigma[bn] * root1[o];
#pragma unroll
    for (int i = 0; i < 16; ++i)
        acc += posenc[node * 16 + i] * root1[(1 + i) * 64 + o];
    acc += agg1[t] / fmaxf(cnt[node], 1.0f);
    xh[t] = (half_t)fmaxf(acc, 0.0f);
}

// -------- layer2 finish: relu(h@root2 + agg/max(cnt,1) + bias2) -> h2 (f32)
__global__ void k_combine2(const half_t* __restrict__ xh, const float* __restrict__ root2,
                           const float* __restrict__ bias2, const float* __restrict__ agg2,
                           const float* __restrict__ cnt, float* __restrict__ h2) {
    int t = blockIdx.x * blockDim.x + threadIdx.x;   // B*N*64 threads
    int o = t & 63, bn = t >> 6;
    int node = bn % NNODE;
    float acc = bias2[o];
    const half_t* row = xh + (size_t)bn * 64;
#pragma unroll 8
    for (int k = 0; k < 64; ++k)
        acc += (float)row[k] * root2[k * 64 + o];
    acc += agg2[t] / fmaxf(cnt[node], 1.0f);
    h2[t] = fmaxf(acc, 0.0f);
}

// ------------------------- heads: mu = mu_pred + h@wm + bm ; r = softplus(h@wr+br)
__global__ void k_final(const float* __restrict__ mu_pred, const float* __restrict__ h2,
                        const float* __restrict__ wm, const float* __restrict__ bm,
                        const float* __restrict__ wr, const float* __restrict__ br,
                        float* __restrict__ out) {
    int t = blockIdx.x * blockDim.x + threadIdx.x;
    if (t >= BB * NNODE) return;
    const float* row = h2 + (size_t)t * 64;
    float dm = 0.0f, dr = 0.0f;
#pragma unroll 8
    for (int k = 0; k < 64; ++k) { dm += row[k] * wm[k]; dr += row[k] * wr[k]; }
    out[t] = mu_pred[t] + dm + bm[0];
    float x = dr + br[0];
    out[BB * NNODE + t] = (x > 20.0f) ? x : log1pf(expf(x));
}

// ---------------------------------------------------------------------------
extern "C" void kernel_launch(void* const* d_in, const int* in_sizes, int n_in,
                              void* d_out, int out_size, void* d_ws, size_t ws_size,
                              hipStream_t stream) {
    (void)in_sizes; (void)n_in; (void)out_size; (void)ws_size;
    const float* mu_pred = (const float*)d_in[0];
    const float* sigma   = (const float*)d_in[1];
    const int*   eidx    = (const int*)d_in[2];
    const float* eattr   = (const float*)d_in[3];
    const float* posenc  = (const float*)d_in[4];
    const float* e1_w1   = (const float*)d_in[5];
    const float* e1_b1   = (const float*)d_in[6];
    const float* e1_w2   = (const float*)d_in[7];
    const float* e1_b2   = (const float*)d_in[8];
    const float* e2_w1   = (const float*)d_in[9];
    const float* e2_b1   = (const float*)d_in[10];
    const float* e2_w2   = (const float*)d_in[11];
    const float* e2_b2   = (const float*)d_in[12];
    const float* root1   = (const float*)d_in[13];
    const float* bias1   = (const float*)d_in[14];
    const float* root2   = (const float*)d_in[15];
    const float* bias2   = (const float*)d_in[16];
    const float* wm      = (const float*)d_in[17];
    const float* bm      = (const float*)d_in[18];
    const float* wr      = (const float*)d_in[19];
    const float* br      = (const float*)d_in[20];
    const int* src = eidx;
    const int* dst = eidx + EE;
    float* out = (float*)d_out;

    // ---- workspace carve (256B aligned) ----
    char* w = (char*)d_ws;
    auto carve = [&](size_t bytes) -> void* {
        void* p = (void*)w;
        w += (bytes + 255) & ~(size_t)255;
        return p;
    };
    half_t* xf     = (half_t*)carve((size_t)BB * NNODE * K1PAD * 2);   // 2.3 MB
    half_t* hid1   = (half_t*)carve((size_t)EE * 64 * 2);              // 1  MB
    half_t* hid2   = (half_t*)carve((size_t)EE * 64 * 2);              // 1  MB
    half_t* w2T1   = (half_t*)carve((size_t)1088 * 64 * 2);
    half_t* w2T2   = (half_t*)carve((size_t)4096 * 64 * 2);
    half_t* weT1   = (half_t*)carve((size_t)EE * 64 * K1PAD * 2);      // 33.5 MB
    half_t* weT2   = (half_t*)carve((size_t)EE * 64 * 64 * 2);         // 67 MB
    float*  agg1   = (float*)carve((size_t)BB * NNODE * 64 * 4);       // 9.3 MB
    float*  agg2   = (float*)carve((size_t)BB * NNODE * 64 * 4);       // 9.3 MB
    half_t* xh     = (half_t*)carve((size_t)BB * NNODE * 64 * 2);      // 4.6 MB
    float*  h2     = (float*)carve((size_t)BB * NNODE * 64 * 4);       // 9.3 MB
    float*  cnt    = (float*)carve((size_t)NNODE * 4);

    const int TB = 256;
    auto blocks = [](size_t n, int tb) { return (unsigned)((n + tb - 1) / tb); };

    // 1) zero accumulators + weT1 pad region (weT1 k=17..31 never stored)
    {
        size_t n = (size_t)EE * 64 * K1PAD / 2;                         // halfs -> u32
        k_zero_u32<<<blocks(n, TB), TB, 0, stream>>>((unsigned int*)weT1, n);
        n = (size_t)BB * NNODE * 64;
        k_zero_u32<<<blocks(n, TB), TB, 0, stream>>>((unsigned int*)agg1, n);
        k_zero_u32<<<blocks(n, TB), TB, 0, stream>>>((unsigned int*)agg2, n);
        k_zero_u32<<<blocks((size_t)NNODE, TB), TB, 0, stream>>>((unsigned int*)cnt, NNODE);
    }

    // 2) stage fp16 operands
    k_build_xf<<<blocks((size_t)BB * NNODE, TB), TB, 0, stream>>>(sigma, posenc, xf);
    k_hidden<<<blocks((size_t)EE * 64, TB), TB, 0, stream>>>(eattr, e1_w1, e1_b1, hid1);
    k_hidden<<<blocks((size_t)EE * 64, TB), TB, 0, stream>>>(eattr, e2_w1, e2_b1, hid2);
    k_cvt_w2T<<<blocks((size_t)1088 * 64, TB), TB, 0, stream>>>(e1_w2, w2T1, 1088);
    k_cvt_w2T<<<blocks((size_t)4096 * 64, TB), TB, 0, stream>>>(e2_w2, w2T2, 4096);

    // 3) edge-MLP second layers on WMMA (E/16 edge tiles x C/16 col chunks)
    k_mlp_wmma<<<dim3(EE / 16, (1088 / 16 + 7) / 8), TB, 0, stream>>>(hid1, w2T1, e1_b2, weT1, 1088, K1PAD);
    k_mlp_wmma<<<dim3(EE / 16, (4096 / 16) / 8), TB, 0, stream>>>(hid2, w2T2, e2_b2, weT2, 4096, 64);

    // 4) degree counts
    k_cnt<<<blocks((size_t)EE, TB), TB, 0, stream>>>(dst, cnt);

    // 5) layer 1: per-edge message GEMM (K=32 padded) + scatter
    k_msg_wmma<K1PAD, 1><<<dim3(EE), TB, 0, stream>>>(xf, weT1, src, dst, agg1);
    k_combine1<<<blocks((size_t)BB * NNODE * 64, TB), TB, 0, stream>>>(sigma, posenc, root1, bias1, agg1, cnt, xh);

    // 6) layer 2: per-edge message GEMM (K=64, 2 k-steps) + scatter
    k_msg_wmma<64, 2><<<dim3(EE), TB, 0, stream>>>(xh, weT2, src, dst, agg2);
    k_combine2<<<blocks((size_t)BB * NNODE * 64, TB), TB, 0, stream>>>(xh, root2, bias2, agg2, cnt, h2);

    // 7) heads
    k_final<<<blocks((size_t)BB * NNODE, TB), TB, 0, stream>>>(mu_pred, h2, wm, bm, wr, br, out);
}
